// ComplexMultiHeadAttention_38027640438991
// MI455X (gfx1250) — matrix operand order repaired and checked by reading, not compiled, not run on hardware
//
#include <hip/hip_runtime.h>
#include <hip/hip_bf16.h>

// ---------------------------------------------------------------------------
// Complex MHA for MI455X (gfx1250, wave32, WMMA + TDM + async-LDS loads).
// All GEMMs run through v_wmma_f32_16x16x32_bf16 (f32 accumulate).
// ---------------------------------------------------------------------------

typedef __bf16 bf16;
typedef bf16  v16bf __attribute__((ext_vector_type(16)));
typedef bf16  v8bf  __attribute__((ext_vector_type(8)));
typedef bf16  v4bf  __attribute__((ext_vector_type(4)));
typedef float v8f   __attribute__((ext_vector_type(8)));
typedef unsigned int u32;
typedef u32   v4u   __attribute__((ext_vector_type(4)));
typedef int   v8i   __attribute__((ext_vector_type(8)));

#define BB   4
#define SS   2048
#define DD   1024
#define HH   16
#define HDIM 64
#define MTOT (BB * SS)   // 8192

// ---------------------------------------------------------------------------

__device__ __forceinline__ v8f wmma_bf16(v16bf a, v16bf b, v8f c) {
  // (neg_a, A, neg_b, B, c_mod, C, reuse_a, reuse_b)
  return __builtin_amdgcn_wmma_f32_16x16x32_bf16(false, a, false, b,
                                                 (short)0, c, false, false);
}

__device__ __forceinline__ v16bf frag_neg(v16bf a) {
  v16bf r;
#pragma unroll
  for (int i = 0; i < 16; i++) r[i] = (bf16)(-(float)a[i]);
  return r;
}

// A-matrix 16x32 bf16 fragment from LDS tile stored [row][k].
// Lanes 0-15: row m=lane, K 0..15; lanes 16-31: same rows, K 16..31
// (matches frag_b_lds). Each group = 8 contiguous halfs -> ds_load_b128.
__device__ __forceinline__ v16bf frag_a_lds(const bf16* base, int stride,
                                            int row, int k, int lane) {
  const int r  = row + (lane & 15);
  const int kk = k + ((lane & 16) ? 16 : 0);
  v8bf lo = *(const v8bf*)(base + r * stride + kk);
  v8bf hi = *(const v8bf*)(base + r * stride + kk + 8);
  v16bf out;
#pragma unroll
  for (int i = 0; i < 8; i++) { out[i] = lo[i]; out[i + 8] = hi[i]; }
  return out;
}

// B-matrix 32x16 bf16 fragment; source tile stored [n][k] so each lane's 16
// K-values are contiguous. Lanes 0-15: col n=lane, K 0..15; lanes 16-31:
// col n=lane-16, K 16..31.
__device__ __forceinline__ v16bf frag_b_lds(const bf16* base, int stride,
                                            int col, int k, int lane) {
  const int n  = col + (lane & 15);
  const int kk = k + ((lane & 16) ? 16 : 0);
  v8bf lo = *(const v8bf*)(base + n * stride + kk);
  v8bf hi = *(const v8bf*)(base + n * stride + kk + 8);
  v16bf out;
#pragma unroll
  for (int i = 0; i < 8; i++) { out[i] = lo[i]; out[i + 8] = hi[i]; }
  return out;
}

// ---------------------------------------------------------------------------
// Tensor Data Mover: 2D bf16 tile load, global -> LDS (D# groups 0/1 only;
// VADDR2/3 = NULL for <=2D tensors). Padding: 4 DWORDs after every 16 DWORDs
// (i.e. rows of 32 halfs land on a 40-half LDS pitch).
// ---------------------------------------------------------------------------
__device__ __forceinline__ void tdm_load_2d(u32 lds_byte_off, const void* gptr,
                                            u32 tile_w, u32 tile_h,
                                            u32 tensor_w, u32 tensor_h,
                                            u32 row_stride) {
  const unsigned long long ga = (unsigned long long)gptr;
  v4u g0;
  g0[0] = 1u;                                       // count=1, user mode
  g0[1] = lds_byte_off;                             // lds_addr
  g0[2] = (u32)ga;                                  // global_addr[31:0]
  g0[3] = (u32)((ga >> 32) & 0x1ffffffu) | (2u << 30);  // addr[56:32] | type=2
  v8i g1;
  g1[0] = (int)((1u << 16)            // data_size = 2 bytes
              | (1u << 20)            // pad_enable
              | (3u << 22)            // pad_interval: 16 DWORDs
              | (3u << 25));          // pad_amount: 4 DWORDs
  g1[1] = (int)((tensor_w & 0xffffu) << 16);                       // dim0 lo
  g1[2] = (int)((tensor_w >> 16) | ((tensor_h & 0xffffu) << 16));  // dim0 hi|dim1 lo
  g1[3] = (int)((tensor_h >> 16) | (tile_w << 16));                // dim1 hi|tile0
  g1[4] = (int)(tile_h & 0xffffu);                                 // tile1 (tile2=0)
  g1[5] = (int)row_stride;                                         // stride0 lo
  g1[6] = 0;
  g1[7] = 0;
  asm volatile("tensor_load_to_lds %0, %1" :: "s"(g0), "s"(g1) : "memory");
}

// per-lane async global -> LDS copy (ASYNCcnt)
__device__ __forceinline__ void async_ld_b128(const bf16* g, const bf16* ldsdst) {
  asm volatile("global_load_async_to_lds_b128 %0, %1, off"
               :: "v"((u32)(unsigned long long)ldsdst),
                  "v"((unsigned long long)g)
               : "memory");
}
__device__ __forceinline__ void wait_async0() {
  asm volatile("s_wait_asynccnt 0" ::: "memory");
}

// ---------------------------------------------------------------------------
// fp32 -> bf16 conversion
// ---------------------------------------------------------------------------
__global__ __launch_bounds__(256)
void cvt_f32_to_bf16(const float* __restrict__ in, bf16* __restrict__ out, int n) {
  const int i = blockIdx.x * 256 + threadIdx.x;
  if (i < n) out[i] = (bf16)in[i];
}

// ---------------------------------------------------------------------------
// Complex GEMM: Yr = Xr Wr^T - Xi Wi^T + br ; Yi = Xr Wi^T + Xi Wr^T + bi
// X: [M=8192, K=1024] bf16; W: [N=1024, K=1024] bf16; 128x64 block tile,
// BK=32, 8 waves (32x32 subtiles). TDM double-buffered staging.
// ---------------------------------------------------------------------------
template <bool OUT_F32>
__global__ __launch_bounds__(256)
void cgemm_kernel(const bf16* __restrict__ Ar, const bf16* __restrict__ Ai,
                  const bf16* __restrict__ Wr, const bf16* __restrict__ Wi,
                  const float* __restrict__ biasR, const float* __restrict__ biasI,
                  void* __restrict__ outR, void* __restrict__ outI) {
  constexpr int K   = DD;
  constexpr int N   = DD;
  constexpr int LDA = 40;  // 32 halfs + TDM pad (4 DWORDs / row)
  __shared__ bf16 sAr[2][128 * LDA], sAi[2][128 * LDA];
  __shared__ bf16 sBr[2][64 * LDA],  sBi[2][64 * LDA];

  const int tid  = threadIdx.x;
  const int wave = tid >> 5, lane = tid & 31;
  const int m0 = blockIdx.y * 128;
  const int n0 = blockIdx.x * 64;
  const int wm = (wave >> 1) * 32;
  const int wn = (wave & 1) * 32;

  const v8f vzero = {0, 0, 0, 0, 0, 0, 0, 0};
  v8f accR[2][2], accI[2][2];
#pragma unroll
  for (int mi = 0; mi < 2; mi++)
#pragma unroll
    for (int ni = 0; ni < 2; ni++) { accR[mi][ni] = vzero; accI[mi][ni] = vzero; }

  auto issue_tiles = [&](int k0, int buf) {
    tdm_load_2d((u32)(unsigned long long)&sAr[buf][0],
                Ar + (size_t)m0 * K + k0, 32, 128, K, MTOT, K);
    tdm_load_2d((u32)(unsigned long long)&sAi[buf][0],
                Ai + (size_t)m0 * K + k0, 32, 128, K, MTOT, K);
    tdm_load_2d((u32)(unsigned long long)&sBr[buf][0],
                Wr + (size_t)n0 * K + k0, 32, 64, K, N, K);
    tdm_load_2d((u32)(unsigned long long)&sBi[buf][0],
                Wi + (size_t)n0 * K + k0, 32, 64, K, N, K);
  };

  if (tid < 32) issue_tiles(0, 0);  // TDM ignores EXEC; one wave drives DMA

  for (int k0 = 0; k0 < K; k0 += 32) {
    const int buf = (k0 >> 5) & 1;
    if (tid < 32) {
      if (k0 + 32 < K) {  // prefetch next tile set into the other buffer
        issue_tiles(k0 + 32, buf ^ 1);
        __builtin_amdgcn_s_wait_tensorcnt(4);  // oldest 4 (current buf) done
      } else {
        __builtin_amdgcn_s_wait_tensorcnt(0);
      }
    }
    __syncthreads();

    const bf16* pAr = sAr[buf]; const bf16* pAi = sAi[buf];
    const bf16* pBr = sBr[buf]; const bf16* pBi = sBi[buf];
    v16bf ar[2], ai[2], ain[2], fbr[2], fbi[2];
#pragma unroll
    for (int mi = 0; mi < 2; mi++) {
      ar[mi]  = frag_a_lds(pAr, LDA, wm + mi * 16, 0, lane);
      ai[mi]  = frag_a_lds(pAi, LDA, wm + mi * 16, 0, lane);
      ain[mi] = frag_neg(ai[mi]);  // bf16 WMMA NEG bits are C-only -> VALU flip
    }
#pragma unroll
    for (int ni = 0; ni < 2; ni++) {
      fbr[ni] = frag_b_lds(pBr, LDA, wn + ni * 16, 0, lane);
      fbi[ni] = frag_b_lds(pBi, LDA, wn + ni * 16, 0, lane);
    }
#pragma unroll
    for (int mi = 0; mi < 2; mi++)
#pragma unroll
      for (int ni = 0; ni < 2; ni++) {
        accR[mi][ni] = wmma_bf16(ar[mi],  fbr[ni], accR[mi][ni]);
        accR[mi][ni] = wmma_bf16(ain[mi], fbi[ni], accR[mi][ni]);
        accI[mi][ni] = wmma_bf16(ar[mi],  fbi[ni], accI[mi][ni]);
        accI[mi][ni] = wmma_bf16(ai[mi],  fbr[ni], accI[mi][ni]);
      }
    __syncthreads();  // all waves done with buf before TDM refills it
  }

  const int half16 = (lane >> 4) & 1;
  const int cl     = lane & 15;
#pragma unroll
  for (int mi = 0; mi < 2; mi++)
#pragma unroll
    for (int ni = 0; ni < 2; ni++) {
      const int col = n0 + wn + ni * 16 + cl;
      const float bR = biasR[col], bI = biasI[col];
#pragma unroll
      for (int j = 0; j < 8; j++) {
        const size_t row = (size_t)(m0 + wm + mi * 16 + j + 8 * half16);
        const size_t off = row * N + col;
        const float vr = accR[mi][ni][j] + bR;
        const float vi = accI[mi][ni][j] + bI;
        if (OUT_F32) {
          ((float*)outR)[off] = vr;
          ((float*)outI)[off] = vi;
        } else {
          ((bf16*)outR)[off] = (bf16)vr;
          ((bf16*)outI)[off] = (bf16)vi;
        }
      }
    }
}

// ---------------------------------------------------------------------------
// RoPE in-place on bf16 Qr,Qi,Kr,Ki stored [B,S,D]; rotates pairs (j, j+32)
// inside each head; angles computed on the fly.
// ---------------------------------------------------------------------------
__global__ __launch_bounds__(256)
void rope_kernel(bf16* __restrict__ qr, bf16* __restrict__ qi,
                 bf16* __restrict__ kr, bf16* __restrict__ ki) {
  const unsigned idx = blockIdx.x * 256 + threadIdx.x;
  const int j = idx & 31;
  unsigned t = idx >> 5;
  const int h = t & (HH - 1);  t >>= 4;
  const int s = t & (SS - 1);  t >>= 11;
  const int b = t;
  const float ang = (float)s * __expf(-(float)j * 0.28782313662425572f);
  float sn, c;
  __sincosf(ang, &sn, &c);
  const size_t base = ((size_t)(b * SS + s)) * DD + h * HDIM + j;
  bf16* ptrs[4] = {qr, qi, kr, ki};
#pragma unroll
  for (int p = 0; p < 4; p++) {
    const float x0 = (float)ptrs[p][base];
    const float x1 = (float)ptrs[p][base + 32];
    ptrs[p][base]      = (bf16)(x0 * c - x1 * sn);
    ptrs[p][base + 32] = (bf16)(x1 * c + x0 * sn);
  }
}

// ---------------------------------------------------------------------------
// Flash attention (real-part scores). One WG per (b, h, 64-query tile),
// 4 waves; wave w owns query rows [w*16, w*16+16). Q/K tiles staged with
// async global->LDS loads; V staged transposed via packed b64 DS stores.
// ---------------------------------------------------------------------------
__global__ __launch_bounds__(128)
void attn_kernel(const bf16* __restrict__ Qr, const bf16* __restrict__ Qi,
                 const bf16* __restrict__ Kr, const bf16* __restrict__ Ki,
                 const bf16* __restrict__ Vr, const bf16* __restrict__ Vi,
                 bf16* __restrict__ Cr, bf16* __restrict__ Ci) {
  constexpr int LDT = 72;  // 64 + 8 halfs pad
  __shared__ bf16 sQr[64 * LDT], sQi[64 * LDT];
  __shared__ bf16 sKr[64 * LDT], sKi[64 * LDT];
  __shared__ bf16 sVr[64 * LDT], sVi[64 * LDT];  // transposed: [d][key]
  __shared__ bf16 sP [64 * LDT];

  const int qt = blockIdx.x, h = blockIdx.y, b = blockIdx.z;
  const int tid = threadIdx.x, wave = tid >> 5, lane = tid & 31;
  const int half16 = (lane >> 4) & 1;

  // ---- stage Q tile via async loads ----
  {
    const int row = tid & 63;
    const int seg = (tid >> 6) * 32;
    const size_t g = ((size_t)(b * SS + qt * 64 + row)) * DD + h * HDIM + seg;
#pragma unroll
    for (int d0 = 0; d0 < 32; d0 += 8) {
      async_ld_b128(Qr + g + d0, sQr + row * LDT + seg + d0);
      async_ld_b128(Qi + g + d0, sQi + row * LDT + seg + d0);
    }
    wait_async0();
  }
  __syncthreads();

  // persistent Q fragments (K-dim 64 = 2 chunks); Qi pre-negated
  v16bf fqr[2], fqin[2];
#pragma unroll
  for (int c = 0; c < 2; c++) {
    fqr[c]  = frag_a_lds(sQr, LDT, wave * 16, c * 32, lane);
    fqin[c] = frag_neg(frag_a_lds(sQi, LDT, wave * 16, c * 32, lane));
  }

  const v8f vzero = {0, 0, 0, 0, 0, 0, 0, 0};
  float m_run[8], l_run[8];
  v8f oR[4], oI[4];
#pragma unroll
  for (int j = 0; j < 8; j++) { m_run[j] = -3.0e30f; l_run[j] = 0.f; }
#pragma unroll
  for (int d = 0; d < 4; d++) { oR[d] = vzero; oI[d] = vzero; }

  for (int kt = 0; kt < SS / 64; kt++) {
    __syncthreads();  // previous iteration done reading K/V/P
    {
      // K tiles: per-lane async global->LDS (no VGPR round trip)
      const int row = tid & 63;
      const int seg = (tid >> 6) * 32;
      const size_t g = ((size_t)(b * SS + kt * 64 + row)) * DD + h * HDIM + seg;
#pragma unroll
      for (int d0 = 0; d0 < 32; d0 += 8) {
        async_ld_b128(Kr + g + d0, sKr + row * LDT + seg + d0);
        async_ld_b128(Ki + g + d0, sKi + row * LDT + seg + d0);
      }
      // V tiles, transposed: 4 keys packed per b64 DS store
      const int kg = (tid & 15) * 4;   // 4 consecutive keys
      const int db = (tid >> 4) * 8;   // 8 dims
      const size_t gv = ((size_t)(b * SS + kt * 64 + kg)) * DD + h * HDIM + db;
      v8bf r0 = *(const v8bf*)(Vr + gv);
      v8bf r1 = *(const v8bf*)(Vr + gv + DD);
      v8bf r2 = *(const v8bf*)(Vr + gv + 2 * DD);
      v8bf r3 = *(const v8bf*)(Vr + gv + 3 * DD);
      v8bf i0 = *(const v8bf*)(Vi + gv);
      v8bf i1 = *(const v8bf*)(Vi + gv + DD);
      v8bf i2 = *(const v8bf*)(Vi + gv + 2 * DD);
      v8bf i3 = *(const v8bf*)(Vi + gv + 3 * DD);
#pragma unroll
      for (int e = 0; e < 8; e++) {
        v4bf pr = {r0[e], r1[e], r2[e], r3[e]};
        v4bf pi = {i0[e], i1[e], i2[e], i3[e]};
        *(v4bf*)(sVr + (db + e) * LDT + kg) = pr;
        *(v4bf*)(sVi + (db + e) * LDT + kg) = pi;
      }
      wait_async0();
    }
    __syncthreads();

    // scores: 16 rows x 64 keys per wave, S = (Qr Kr^T - Qi Ki^T)/8
    v8f sc[4];
#pragma unroll
    for (int nf = 0; nf < 4; nf++) {
      v8f a = vzero;
#pragma unroll
      for (int c = 0; c < 2; c++) {
        a = wmma_bf16(fqr[c],  frag_b_lds(sKr, LDT, nf * 16, c * 32, lane), a);
        a = wmma_bf16(fqin[c], frag_b_lds(sKi, LDT, nf * 16, c * 32, lane), a);
      }
#pragma unroll
      for (int j = 0; j < 8; j++) a[j] *= 0.125f;
      sc[nf] = a;
    }

    // online softmax; row stats per (VGPR j, lane half), reduced over the 16
    // lanes sharing a row with width-16 xor shuffles
    float fac[8], rsum[8];
#pragma unroll
    for (int j = 0; j < 8; j++) {
      float v = fmaxf(fmaxf(sc[0][j], sc[1][j]), fmaxf(sc[2][j], sc[3][j]));
#pragma unroll
      for (int off = 1; off < 16; off <<= 1)
        v = fmaxf(v, __shfl_xor(v, off, 16));
      const float mn = fmaxf(m_run[j], v);
      fac[j]  = __expf(m_run[j] - mn);
      m_run[j] = mn;
      rsum[j] = 0.f;
    }
#pragma unroll
    for (int nf = 0; nf < 4; nf++)
#pragma unroll
      for (int j = 0; j < 8; j++) {
        const float p = __expf(sc[nf][j] - m_run[j]);
        rsum[j] += p;
        sP[(wave * 16 + j + 8 * half16) * LDT + nf * 16 + (lane & 15)] = (bf16)p;
      }
#pragma unroll
    for (int j = 0; j < 8; j++) {
#pragma unroll
      for (int off = 1; off < 16; off <<= 1)
        rsum[j] += __shfl_xor(rsum[j], off, 16);
      l_run[j] = l_run[j] * fac[j] + rsum[j];
    }
#pragma unroll
    for (int d = 0; d < 4; d++)
#pragma unroll
      for (int j = 0; j < 8; j++) { oR[d][j] *= fac[j]; oI[d][j] *= fac[j]; }

    __syncthreads();  // P visible

    // O += P @ V (contraction over 64 keys = 2 chunks)
#pragma unroll
    for (int c = 0; c < 2; c++) {
      v16bf ap = frag_a_lds(sP, LDT, wave * 16, c * 32, lane);
#pragma unroll
      for (int d = 0; d < 4; d++) {
        oR[d] = wmma_bf16(ap, frag_b_lds(sVr, LDT, d * 16, c * 32, lane), oR[d]);
        oI[d] = wmma_bf16(ap, frag_b_lds(sVi, LDT, d * 16, c * 32, lane), oI[d]);
      }
    }
  }

  // normalize + write context (bf16), merged [B,S,D]
  float inv[8];
#pragma unroll
  for (int j = 0; j < 8; j++) inv[j] = 1.0f / l_run[j];
#pragma unroll
  for (int d = 0; d < 4; d++) {
    const int col = h * HDIM + d * 16 + (lane & 15);
#pragma unroll
    for (int j = 0; j < 8; j++) {
      const int srow = qt * 64 + wave * 16 + j + 8 * half16;
      const size_t g = ((size_t)(b * SS + srow)) * DD + col;
      Cr[g] = (bf16)(oR[d][j] * inv[j]);
      Ci[g] = (bf16)(oI[d][j] * inv[j]);
    }
  }
}

// ---------------------------------------------------------------------------
// Host orchestration
// ---------------------------------------------------------------------------
extern "C" void kernel_launch(void* const* d_in, const int* in_sizes, int n_in,
                              void* d_out, int out_size, void* d_ws, size_t ws_size,
                              hipStream_t stream) {
  (void)in_sizes; (void)n_in; (void)out_size; (void)ws_size;

  const float* x_r = (const float*)d_in[0];
  const float* x_i = (const float*)d_in[1];
  const float* wsrc[8] = {(const float*)d_in[2], (const float*)d_in[3],
                          (const float*)d_in[4], (const float*)d_in[5],
                          (const float*)d_in[6], (const float*)d_in[7],
                          (const float*)d_in[8], (const float*)d_in[9]};
  const float* bq_r = (const float*)d_in[10]; const float* bq_i = (const float*)d_in[11];
  const float* bk_r = (const float*)d_in[12]; const float* bk_i = (const float*)d_in[13];
  const float* bv_r = (const float*)d_in[14]; const float* bv_i = (const float*)d_in[15];
  const float* bo_r = (const float*)d_in[16]; const float* bo_i = (const float*)d_in[17];

  const size_t MD = (size_t)MTOT * DD;
  const size_t WW = (size_t)DD * DD;
  bf16* p = (bf16*)d_ws;
  bf16* xr16 = p; p += MD;
  bf16* xi16 = p; p += MD;
  bf16* w16[8];
  for (int i = 0; i < 8; i++) { w16[i] = p; p += WW; }
  bf16* qr16 = p; p += MD;  bf16* qi16 = p; p += MD;
  bf16* kr16 = p; p += MD;  bf16* ki16 = p; p += MD;
  bf16* vr16 = p; p += MD;  bf16* vi16 = p; p += MD;
  bf16* cr16 = p; p += MD;  bf16* ci16 = p; p += MD;

  cvt_f32_to_bf16<<<(int)(MD / 256), 256, 0, stream>>>(x_r, xr16, (int)MD);
  cvt_f32_to_bf16<<<(int)(MD / 256), 256, 0, stream>>>(x_i, xi16, (int)MD);
  for (int i = 0; i < 8; i++)
    cvt_f32_to_bf16<<<(int)(WW / 256), 256, 0, stream>>>(wsrc[i], w16[i], (int)WW);

  dim3 gg(DD / 64, MTOT / 128);
  cgemm_kernel<false><<<gg, 256, 0, stream>>>(xr16, xi16, w16[0], w16[1],
                                              bq_r, bq_i, qr16, qi16);
  cgemm_kernel<false><<<gg, 256, 0, stream>>>(xr16, xi16, w16[2], w16[3],
                                              bk_r, bk_i, kr16, ki16);
  cgemm_kernel<false><<<gg, 256, 0, stream>>>(xr16, xi16, w16[4], w16[5],
                                              bv_r, bv_i, vr16, vi16);

  rope_kernel<<<(BB * SS * HH * 32) / 256, 256, 0, stream>>>(qr16, qi16, kr16, ki16);

  attn_kernel<<<dim3(SS / 64, HH, BB), 128, 0, stream>>>(qr16, qi16, kr16, ki16,
                                                         vr16, vi16, cr16, ci16);

  float* outR = (float*)d_out;
  float* outI = outR + MD;
  cgemm_kernel<true><<<gg, 256, 0, stream>>>(cr16, ci16, w16[6], w16[7],
                                             bo_r, bo_i, outR, outI);
}
